// resPromptClip_10608569221227
// MI455X (gfx1250) — matrix operand order repaired and checked by reading, not compiled, not run on hardware
//
#include <hip/hip_runtime.h>

// ---------------- problem constants ----------------
#define C_DIM 768
#define NH    12
#define HD    64
#define HIDN  3072
#define BSZ   4
#define TT    8
#define HWP   196          // 14*14
#define N1    1568         // tokens per batch w/o cls
#define NTOT  1569
#define MT    6272         // BSZ*N1   temporal rows
#define SEQ   197          // spatial sequence (1+196)
#define SEQP  208          // padded to 16 (M/N of attention)
#define SEQK  224          // padded to 32 (K dim of P.V)
#define MS    6304         // 32*197   spatial rows
#define MX    6276         // BSZ*NTOT
#define BHS   384          // 32*12 batched attention heads

typedef unsigned short u16;
typedef __bf16  bf16_t;
typedef __bf16  v8bf  __attribute__((ext_vector_type(8)));
typedef __bf16  v16bf __attribute__((ext_vector_type(16)));
typedef float   v8f   __attribute__((ext_vector_type(8)));

__device__ __forceinline__ u16 f32_bf16(float f) {
  union { float f; unsigned u; } v; v.f = f;
  unsigned u = v.u;
  u += 0x7FFFu + ((u >> 16) & 1u);   // RNE
  return (u16)(u >> 16);
}

// ---------------- fp32 -> bf16 weight conversion ----------------
__global__ __launch_bounds__(256)
void cvt_bf16_kernel(const float* __restrict__ in, u16* __restrict__ out, int n) {
  int i = blockIdx.x * 256 + threadIdx.x;
  if (i < n) out[i] = f32_bf16(in[i]);
}

// ---------------- layer norm (3 source-mapping modes) -> bf16 rows ----------------
__global__ __launch_bounds__(256)
void ln_kernel(const float* __restrict__ srcA, const float* __restrict__ srcB,
               const float* __restrict__ gg, const float* __restrict__ bb,
               u16* __restrict__ out, int mode)
{
  int row = blockIdx.x;
  const float* src;
  if (mode == 0) {
    int b = row / N1;
    src = srcA + (size_t)(row + b + 1) * C_DIM;
  } else if (mode == 1) {
    int bt = row / SEQ, pos = row - bt * SEQ;
    int b = bt >> 3, t = bt & 7;
    if (pos == 0) src = srcA + (size_t)(b * NTOT) * C_DIM;
    else          src = srcB + (size_t)(b * N1 + (pos - 1) * TT + t) * C_DIM;
  } else {
    src = srcA + (size_t)row * C_DIM;
  }
  int tid = threadIdx.x;
  float x0 = src[tid], x1 = src[tid + 256], x2 = src[tid + 512];
  __shared__ float s1[256], s2[256];
  s1[tid] = x0 + x1 + x2;
  s2[tid] = x0 * x0 + x1 * x1 + x2 * x2;
  __syncthreads();
  for (int off = 128; off > 0; off >>= 1) {
    if (tid < off) { s1[tid] += s1[tid + off]; s2[tid] += s2[tid + off]; }
    __syncthreads();
  }
  float mu   = s1[0] * (1.f / 768.f);
  float var  = s2[0] * (1.f / 768.f) - mu * mu;
  float rstd = rsqrtf(var + 1e-5f);
  u16* o = out + (size_t)row * C_DIM;
  o[tid      ] = f32_bf16((x0 - mu) * rstd * gg[tid      ] + bb[tid      ]);
  o[tid + 256] = f32_bf16((x1 - mu) * rstd * gg[tid + 256] + bb[tid + 256]);
  o[tid + 512] = f32_bf16((x2 - mu) * rstd * gg[tid + 512] + bb[tid + 512]);
}

// ---------------- generic batched WMMA GEMM: out = A(MxK) * Bw(NxK)^T ----------------
// LDS double-buffered pipeline fed by global_load_async_to_lds_b128 (ASYNCcnt).
// A, Bw: bf16 row-major (lda = ldb = K). Epilogue: +bias, gelu, +resid, f32/bf16 out.
template<int NT>
__global__ __launch_bounds__(256)
void gemm_wmma_kernel(const u16* __restrict__ A, const u16* __restrict__ Bw,
                      const float* __restrict__ bias,
                      float* __restrict__ outF, u16* __restrict__ outH,
                      const float* __restrict__ resid,
                      int M, int N, int K,
                      long long sA, long long sB, long long sC, int actGelu)
{
  constexpr int BN   = NT * 64;   // block N tile
  constexpr int ROWP = 40;        // 32 data + 8 pad (u16): 20-word bank stride, 16B aligned
  __shared__ alignas(32) u16 As[2][32][ROWP];
  __shared__ alignas(32) u16 Bs[2][BN][ROWP];

  const u16* Ab = A + (size_t)blockIdx.z * sA;
  const u16* Bb = Bw + (size_t)blockIdx.z * sB;
  size_t coff = (size_t)blockIdx.z * sC;
  int tid  = threadIdx.x;
  int lane = tid & 31;
  int wid  = tid >> 5;
  int wr = wid >> 2, wc = wid & 3;                 // 2 x 4 waves
  int mTile = blockIdx.x * 32;
  int nTile = blockIdx.y * BN;
  int ml = lane & 15;                              // tile row/col index
  int kh = lane >> 4;                              // K-half selector

  unsigned ldsA = (unsigned)(size_t)(void*)&As[0][0][0];
  unsigned ldsB = (unsigned)(size_t)(void*)&Bs[0][0][0];

  int segr = tid >> 2;                             // 0..63 : copy row
  int sege = (tid & 3) * 8;                        // 0,8,16,24 : element offset (16B seg)

  auto issue_stage = [&](int k0, int buf) {
    if (tid < 128) {                               // A tile: 32 rows x 4 segs (wave-uniform)
      int r = tid >> 2;
      int gm = mTile + r; if (gm > M - 1) gm = M - 1;
      unsigned long long ga =
          (unsigned long long)(size_t)(Ab + (size_t)gm * K + k0 + sege);
      unsigned la = ldsA + (unsigned)(((buf * 32 + r) * ROWP + sege) * 2);
      asm volatile("global_load_async_to_lds_b128 %0, %1, off"
                   :: "v"(la), "v"(ga) : "memory");
    }
#pragma unroll
    for (int it = 0; it < NT; ++it) {              // B tile: BN rows x 4 segs
      int rr = segr + it * 64;
      int gn = nTile + rr; if (gn > N - 1) gn = N - 1;
      unsigned long long gb =
          (unsigned long long)(size_t)(Bb + (size_t)gn * K + k0 + sege);
      unsigned lb = ldsB + (unsigned)(((buf * BN + rr) * ROWP + sege) * 2);
      asm volatile("global_load_async_to_lds_b128 %0, %1, off"
                   :: "v"(lb), "v"(gb) : "memory");
    }
  };

  v8f acc[NT];
#pragma unroll
  for (int j = 0; j < NT; ++j)
#pragma unroll
    for (int e = 0; e < 8; ++e) acc[j][e] = 0.f;

  issue_stage(0, 0);
  int nK = K >> 5;
  for (int ks = 0; ks < nK; ++ks) {
    asm volatile("s_wait_asynccnt 0x0" ::: "memory");  // own copies of stage ks done
    __syncthreads();                                   // whole tile visible; prev reads done
    if (ks + 1 < nK) issue_stage((ks + 1) << 5, (ks + 1) & 1);
    int cur = ks & 1;
    const u16* ar = &As[cur][wr * 16 + ml][kh * 8];
    v8bf alo = *(const v8bf*)(ar);
    v8bf ahi = *(const v8bf*)(ar + 16);
    v16bf af = __builtin_shufflevector(alo, ahi,
        0, 1, 2, 3, 4, 5, 6, 7, 8, 9, 10, 11, 12, 13, 14, 15);
#pragma unroll
    for (int j = 0; j < NT; ++j) {
      const u16* br = &Bs[cur][wc * (NT * 16) + j * 16 + ml][kh * 8];
      v8bf blo = *(const v8bf*)(br);
      v8bf bhi = *(const v8bf*)(br + 16);
      v16bf bv = __builtin_shufflevector(blo, bhi,
          0, 1, 2, 3, 4, 5, 6, 7, 8, 9, 10, 11, 12, 13, 14, 15);
      acc[j] = __builtin_amdgcn_wmma_f32_16x16x32_bf16(
          false, af, false, bv, (short)0, acc[j], false, false);
    }
  }

#pragma unroll
  for (int j = 0; j < NT; ++j) {
    int n = nTile + wc * (NT * 16) + j * 16 + ml;
    if (n >= N) continue;
    float bv = bias ? bias[n] : 0.f;
#pragma unroll
    for (int r = 0; r < 8; ++r) {
      int m = mTile + wr * 16 + r + (kh << 3);
      if (m >= M) continue;
      float v = acc[j][r] + bv;
      if (actGelu) v = 0.5f * v * (1.f + erff(v * 0.70710678118f));
      size_t idx = coff + (size_t)m * N + n;
      if (resid) v += resid[idx];
      if (outF) outF[idx] = v;
      if (outH) outH[idx] = f32_bf16(v);
    }
  }
}

// ---------------- temporal attention (seq=8, D=64) : one block per (group, head) ----------------
__global__ __launch_bounds__(64)
void attn_t_kernel(const float* __restrict__ qkv, u16* __restrict__ out)
{
  int g = blockIdx.x;                // 0..783  (b*196+hw)
  int h = blockIdx.y;                // 0..11
  __shared__ float Q[8][64], Kx[8][64], V[8][64], S[8][8], P[8][8];
  int tid = threadIdx.x;
#pragma unroll
  for (int j = 0; j < 8; ++j) {
    const float* r = qkv + ((size_t)(g * 8 + j) * (3 * C_DIM)) + h * HD;
    Q[j][tid]  = r[tid];
    Kx[j][tid] = r[C_DIM + tid];
    V[j][tid]  = r[2 * C_DIM + tid];
  }
  __syncthreads();
  {
    int t = tid >> 3, j = tid & 7;
    float s = 0.f;
#pragma unroll
    for (int d = 0; d < 64; ++d) s += Q[t][d] * Kx[j][d];
    S[t][j] = s * 0.125f;
  }
  __syncthreads();
  if (tid < 8) {
    float m = -3.0e38f;
#pragma unroll
    for (int j = 0; j < 8; ++j) m = fmaxf(m, S[tid][j]);
    float sum = 0.f;
    float e[8];
#pragma unroll
    for (int j = 0; j < 8; ++j) { e[j] = __expf(S[tid][j] - m); sum += e[j]; }
    float inv = 1.f / sum;
#pragma unroll
    for (int j = 0; j < 8; ++j) P[tid][j] = e[j] * inv;
  }
  __syncthreads();
  {
    int t = tid >> 3, d0 = (tid & 7) * 8;
#pragma unroll
    for (int dd = 0; dd < 8; ++dd) {
      float o = 0.f;
#pragma unroll
      for (int j = 0; j < 8; ++j) o += P[t][j] * V[j][d0 + dd];
      out[((size_t)(g * 8 + t)) * C_DIM + h * HD + d0 + dd] = f32_bf16(o);
    }
  }
}

// ---------------- xt = x[:,1:,:] + t_fc output ----------------
__global__ __launch_bounds__(256)
void add_resid_kernel(const float* __restrict__ x, const float* __restrict__ z,
                      float* __restrict__ xt)
{
  size_t idx = (size_t)blockIdx.x * 256 + threadIdx.x;
  if (idx >= (size_t)MT * C_DIM) return;
  int row = (int)(idx / C_DIM);
  int c   = (int)(idx - (size_t)row * C_DIM);
  int b   = row / N1;
  xt[idx] = x[(size_t)(row + b + 1) * C_DIM + c] + z[idx];
}

// ---------------- repack spatial qkv -> padded Q(208x64), K(208x64), Vt(64x224) per (bt,h) ----------------
__global__ __launch_bounds__(256)
void repack_qkv_kernel(const float* __restrict__ qkv,
                       u16* __restrict__ Qo, u16* __restrict__ Ko, u16* __restrict__ Vt)
{
  int bh = blockIdx.x;               // 0..383
  int bt = bh / NH, h = bh - bt * NH;
  const float* base = qkv + (size_t)bt * SEQ * (3 * C_DIM) + h * HD;
  u16* q = Qo + (size_t)bh * SEQP * HD;
  u16* k = Ko + (size_t)bh * SEQP * HD;
  for (int i = threadIdx.x; i < SEQP * HD; i += 256) {
    int pos = i >> 6, d = i & 63;
    float qv = 0.f, kv = 0.f;
    if (pos < SEQ) {
      const float* r = base + (size_t)pos * (3 * C_DIM);
      qv = r[d] * 0.125f;            // pre-scale by 1/sqrt(64)
      kv = r[C_DIM + d];
    }
    q[i] = f32_bf16(qv);
    k[i] = f32_bf16(kv);
  }
  u16* vt = Vt + (size_t)bh * HD * SEQK;
  for (int i = threadIdx.x; i < HD * SEQK; i += 256) {
    int d = i / SEQK, j = i - d * SEQK;
    float vv = (j < SEQ) ? base[(size_t)j * (3 * C_DIM) + 2 * C_DIM + d] : 0.f;
    vt[i] = f32_bf16(vv);
  }
}

// ---------------- masked row softmax: S(208x208 f32) -> P(208x224 bf16) ----------------
__global__ __launch_bounds__(64)
void softmax_kernel(const float* __restrict__ S, u16* __restrict__ P)
{
  int i  = blockIdx.x;               // row 0..207
  int bh = blockIdx.y;
  int tid = threadIdx.x;
  const float* s = S + ((size_t)bh * SEQP + i) * SEQP;
  u16* p = P + ((size_t)bh * SEQP + i) * SEQK;
  if (i >= SEQ) {                    // padded rows -> zero
    for (int j = tid; j < SEQK; j += 64) p[j] = 0;
    return;
  }
  __shared__ float red[64];
  float mx = -3.0e38f;
  for (int j = tid; j < SEQ; j += 64) mx = fmaxf(mx, s[j]);
  red[tid] = mx; __syncthreads();
  for (int off = 32; off > 0; off >>= 1) {
    if (tid < off) red[tid] = fmaxf(red[tid], red[tid + off]);
    __syncthreads();
  }
  mx = red[0]; __syncthreads();
  float sum = 0.f;
  for (int j = tid; j < SEQ; j += 64) sum += __expf(s[j] - mx);
  red[tid] = sum; __syncthreads();
  for (int off = 32; off > 0; off >>= 1) {
    if (tid < off) red[tid] += red[tid + off];
    __syncthreads();
  }
  float inv = 1.f / red[0];
  for (int j = tid; j < SEQK; j += 64)
    p[j] = (j < SEQ) ? f32_bf16(__expf(s[j] - mx) * inv) : (u16)0;
}

// ---------------- scatter attention output back to token rows ----------------
__global__ __launch_bounds__(256)
void repack_o_kernel(const u16* __restrict__ Os, u16* __restrict__ Ao)
{
  int bh = blockIdx.x;
  int bt = bh / NH, h = bh - bt * NH;
  const u16* o = Os + (size_t)bh * SEQP * HD;
  for (int i = threadIdx.x; i < SEQ * HD; i += 256) {
    int pos = i >> 6, d = i & 63;
    Ao[((size_t)(bt * SEQ + pos)) * C_DIM + h * HD + d] = o[pos * HD + d];
  }
}

// ---------------- final residual combine incl. cls mean ----------------
__global__ __launch_bounds__(256)
void combine_kernel(const float* __restrict__ x, const float* __restrict__ xt,
                    const float* __restrict__ Rs, float* __restrict__ xnew)
{
  size_t idx = (size_t)blockIdx.x * 256 + threadIdx.x;
  if (idx >= (size_t)MX * C_DIM) return;
  int c = (int)(idx % C_DIM);
  int n = (int)(idx / C_DIM);
  int b = n / NTOT, p = n - b * NTOT;
  float v;
  if (p == 0) {
    float s = 0.f;
#pragma unroll
    for (int t = 0; t < TT; ++t)
      s += Rs[((size_t)((b * TT + t) * SEQ)) * C_DIM + c];
    v = x[idx] + s * 0.125f;
  } else {
    int n1 = p - 1, hw = n1 >> 3, t = n1 & 7;
    v = xt[((size_t)(b * N1 + n1)) * C_DIM + c]
      + Rs[((size_t)((b * TT + t) * SEQ + 1 + hw)) * C_DIM + c];
  }
  xnew[idx] = v;
}

// =======================================================================
extern "C" void kernel_launch(void* const* d_in, const int* in_sizes, int n_in,
                              void* d_out, int out_size, void* d_ws, size_t ws_size,
                              hipStream_t stream)
{
  (void)in_sizes; (void)n_in; (void)out_size; (void)ws_size;
  const float* x        = (const float*)d_in[0];
  const float* ln_t_g   = (const float*)d_in[1];
  const float* ln_t_b   = (const float*)d_in[2];
  const float* t_qkv_w  = (const float*)d_in[3];
  const float* t_qkv_b  = (const float*)d_in[4];
  const float* t_proj_w = (const float*)d_in[5];
  const float* t_proj_b = (const float*)d_in[6];
  const float* t_fc_w   = (const float*)d_in[7];
  const float* t_fc_b   = (const float*)d_in[8];
  const float* ln1_g    = (const float*)d_in[9];
  const float* ln1_b    = (const float*)d_in[10];
  const float* s_qkv_w  = (const float*)d_in[11];
  const float* s_qkv_b  = (const float*)d_in[12];
  const float* s_proj_w = (const float*)d_in[13];
  const float* s_proj_b = (const float*)d_in[14];
  const float* ln2_g    = (const float*)d_in[15];
  const float* ln2_b    = (const float*)d_in[16];
  const float* fc1_w    = (const float*)d_in[17];
  const float* fc1_b    = (const float*)d_in[18];
  const float* fc2_w    = (const float*)d_in[19];
  const float* fc2_b    = (const float*)d_in[20];
  float* out = (float*)d_out;

  // ---- bump allocator over workspace (256B aligned) ----
  char* ws = (char*)d_ws;
  size_t off = 0;
  auto alloc = [&](size_t bytes) -> char* {
    char* p = ws + off;
    off += (bytes + 255) & ~(size_t)255;
    return p;
  };
  u16* wb_tqkv  = (u16*)alloc((size_t)2304 * 768 * 2);
  u16* wb_tproj = (u16*)alloc((size_t)768 * 768 * 2);
  u16* wb_tfc   = (u16*)alloc((size_t)768 * 768 * 2);
  u16* wb_sqkv  = (u16*)alloc((size_t)2304 * 768 * 2);
  u16* wb_sproj = (u16*)alloc((size_t)768 * 768 * 2);
  u16* wb_fc1   = (u16*)alloc((size_t)HIDN * 768 * 2);
  u16* wb_fc2   = (u16*)alloc((size_t)768 * HIDN * 2);
  float* region1 = (float*)alloc((size_t)BHS * SEQP * SEQP * 4); // qkv_t / qkv_s / S
  u16*  bufA = (u16*)alloc((size_t)MS * C_DIM * 2);              // A_t -> O_t -> A_o
  u16*  bufB = (u16*)alloc((size_t)MS * C_DIM * 2);              // Z -> A_s -> A_m
  float* bufF = (float*)alloc((size_t)MS * C_DIM * 4);           // Zfc -> R_s
  float* xt   = (float*)alloc((size_t)MT * C_DIM * 4);
  u16* Qb  = (u16*)alloc((size_t)BHS * SEQP * HD * 2);
  u16* Kb  = (u16*)alloc((size_t)BHS * SEQP * HD * 2);
  u16* Vtb = (u16*)alloc((size_t)BHS * HD * SEQK * 2);
  u16* Pb  = (u16*)alloc((size_t)BHS * SEQP * SEQK * 2);
  u16* Osb = (u16*)alloc((size_t)BHS * SEQP * HD * 2);
  float* xnew = (float*)alloc((size_t)MX * C_DIM * 4);
  u16* Ah = (u16*)alloc((size_t)MX * HIDN * 2);

  auto cvt = [&](const float* in, u16* o, size_t n) {
    cvt_bf16_kernel<<<dim3((unsigned)((n + 255) / 256)), dim3(256), 0, stream>>>(in, o, (int)n);
  };
  cvt(t_qkv_w,  wb_tqkv,  (size_t)2304 * 768);
  cvt(t_proj_w, wb_tproj, (size_t)768 * 768);
  cvt(t_fc_w,   wb_tfc,   (size_t)768 * 768);
  cvt(s_qkv_w,  wb_sqkv,  (size_t)2304 * 768);
  cvt(s_proj_w, wb_sproj, (size_t)768 * 768);
  cvt(fc1_w,    wb_fc1,   (size_t)HIDN * 768);
  cvt(fc2_w,    wb_fc2,   (size_t)768 * HIDN);

  auto gemm4 = [&](const u16* A, const u16* Bm, const float* bias,
                   float* oF, u16* oH, const float* resid,
                   int M, int N, int K, int gelu) {
    dim3 g((M + 31) / 32, (N + 255) / 256, 1);
    gemm_wmma_kernel<4><<<g, 256, 0, stream>>>(A, Bm, bias, oF, oH, resid,
                                               M, N, K, 0, 0, 0, gelu);
  };

  // ---- temporal branch ----
  ln_kernel<<<MT, 256, 0, stream>>>(x, nullptr, ln_t_g, ln_t_b, bufA, 0);
  gemm4(bufA, wb_tqkv, t_qkv_b, region1, nullptr, nullptr, MT, 2304, 768, 0);
  attn_t_kernel<<<dim3(BSZ * HWP, NH), 64, 0, stream>>>(region1, bufA);
  gemm4(bufA, wb_tproj, t_proj_b, nullptr, bufB, nullptr, MT, 768, 768, 0);   // Z
  gemm4(bufB, wb_tfc, t_fc_b, bufF, nullptr, nullptr, MT, 768, 768, 0);       // Zfc
  add_resid_kernel<<<(MT * C_DIM + 255) / 256, 256, 0, stream>>>(x, bufF, xt);

  // ---- spatial branch ----
  ln_kernel<<<MS, 256, 0, stream>>>(x, xt, ln1_g, ln1_b, bufB, 1);
  gemm4(bufB, wb_sqkv, s_qkv_b, region1, nullptr, nullptr, MS, 2304, 768, 0); // qkv_s
  repack_qkv_kernel<<<BHS, 256, 0, stream>>>(region1, Qb, Kb, Vtb);
  {   // S = Q K^T  (batched over 384 heads)
    dim3 g((SEQP + 31) / 32, (SEQP + 255) / 256, BHS);
    gemm_wmma_kernel<4><<<g, 256, 0, stream>>>(Qb, Kb, nullptr, region1, nullptr, nullptr,
        SEQP, SEQP, HD,
        (long long)SEQP * HD, (long long)SEQP * HD, (long long)SEQP * SEQP, 0);
  }
  softmax_kernel<<<dim3(SEQP, BHS), 64, 0, stream>>>(region1, Pb);
  {   // O = P V  (batched; B matrix is V^T stored 64x224)
    dim3 g((SEQP + 31) / 32, (HD + 63) / 64, BHS);
    gemm_wmma_kernel<1><<<g, 256, 0, stream>>>(Pb, Vtb, nullptr, nullptr, Osb, nullptr,
        SEQP, HD, SEQK,
        (long long)SEQP * SEQK, (long long)HD * SEQK, (long long)SEQP * HD, 0);
  }
  repack_o_kernel<<<BHS, 256, 0, stream>>>(Osb, bufA);
  gemm4(bufA, wb_sproj, s_proj_b, bufF, nullptr, nullptr, MS, 768, 768, 0);   // R_s
  combine_kernel<<<(MX * C_DIM + 255) / 256, 256, 0, stream>>>(x, xt, bufF, xnew);

  // ---- MLP ----
  ln_kernel<<<MX, 256, 0, stream>>>(xnew, nullptr, ln2_g, ln2_b, bufB, 2);
  gemm4(bufB, wb_fc1, fc1_b, nullptr, Ah, nullptr, MX, HIDN, 768, 1);         // gelu -> bf16
  gemm4(Ah, wb_fc2, fc2_b, out, nullptr, xnew, MX, 768, HIDN, 0);             // + residual
}
